// GLALayer_2499670966947
// MI455X (gfx1250) — matrix operand order repaired and checked by reading, not compiled, not run on hardware
//
#include <hip/hip_runtime.h>
#include <hip/hip_bf16.h>
#include <math.h>

// ---------------------------------------------------------------------------
// Types for CDNA5 WMMA
// ---------------------------------------------------------------------------
typedef __attribute__((ext_vector_type(16))) __bf16          v16bf;
typedef __attribute__((ext_vector_type(8)))  float           v8f;
typedef __attribute__((ext_vector_type(8)))  unsigned short  us8;
typedef __attribute__((ext_vector_type(4)))  float           f4;

#define D_MODEL 1024
#define NHEAD   16
#define DHEAD   64
#define FFN_DIM 4096
#define CHUNK   64
#define SEQ_N   4096
#define BATCH   2
#define TOKENS  (BATCH * SEQ_N)

// Async global->LDS path (gfx1250 GLOBAL_LOAD_ASYNC_TO_LDS_B128, ASYNCcnt)
#if defined(__AMDGCN__) && __has_builtin(__builtin_amdgcn_global_load_async_to_lds_b128) && \
    __has_builtin(__builtin_amdgcn_s_wait_asynccnt)
#define HAVE_ASYNC_LDS 1
#else
#define HAVE_ASYNC_LDS 0
#endif

#if HAVE_ASYNC_LDS
typedef int i4v __attribute__((vector_size(16)));
__device__ __forceinline__ void async_cp16(unsigned short* dst_lds,
                                           const unsigned short* src) {
    __builtin_amdgcn_global_load_async_to_lds_b128(
        (__attribute__((address_space(1))) i4v*)(src),
        (__attribute__((address_space(3))) i4v*)(dst_lds), 0, 0);
}
#endif

// ---------------------------------------------------------------------------
// bf16 helpers
// ---------------------------------------------------------------------------
__device__ __forceinline__ unsigned short f2b(float f) {
    unsigned int u = __float_as_uint(f);
    u += 0x7FFFu + ((u >> 16) & 1u);
    return (unsigned short)(u >> 16);
}
__device__ __forceinline__ float b2f(unsigned short h) {
    return __uint_as_float(((unsigned int)h) << 16);
}

__device__ __forceinline__ v16bf make_frag(us8 lo, us8 hi) {
    union { us8 u[2]; v16bf v; } c;
    c.u[0] = lo; c.u[1] = hi;
    return c.v;
}

// ---------------------------------------------------------------------------
// WMMA fragment loaders (wave32, v_wmma_f32_16x16x32_bf16, ISA 7.12.2 layouts)
//   A 16x32: lane m=lane&15; elem j -> K = ((j<8)? j : j+8) + (lane&16 ? 8:0)
//   B 32x16: lane n=lane&15; elem j -> K = ((j<8)? j : j+8) + (lane&16 ? 8:0)
//   C/D:     lane n=lane&15; vgpr r -> M = r + (lane&16 ? 8:0)
// Leading dims are multiples of 8 ushorts (16B) -> all loads are ds_load_b128.
// ---------------------------------------------------------------------------
__device__ __forceinline__ v16bf load_a_vec(const unsigned short* s, int ld, int lane) {
    const unsigned short* p = s + (lane & 15) * ld + ((lane & 16) ? 8 : 0);
    return make_frag(*(const us8*)p, *(const us8*)(p + 16));
}
__device__ __forceinline__ v16bf load_b_vecT(const unsigned short* sT, int ld, int lane) {
    const unsigned short* p = sT + (lane & 15) * ld + ((lane & 16) ? 8 : 0);
    return make_frag(*(const us8*)p, *(const us8*)(p + 16));
}

__device__ __forceinline__ v8f wmma_bf16(v16bf a, v16bf b, v8f c) {
    return __builtin_amdgcn_wmma_f32_16x16x32_bf16(false, a, false, b, (short)0, c,
                                                   false, false);
}

// ---------------------------------------------------------------------------
// fp32 -> bf16 conversion (weights)
// ---------------------------------------------------------------------------
__global__ void cvt_f32_bf16(const float* __restrict__ in,
                             unsigned short* __restrict__ out, size_t n) {
    for (size_t i = blockIdx.x * (size_t)blockDim.x + threadIdx.x; i < n;
         i += (size_t)gridDim.x * blockDim.x)
        out[i] = f2b(in[i]);
}

// ---------------------------------------------------------------------------
// LayerNorm (row = 1024) -> bf16
// ---------------------------------------------------------------------------
__global__ __launch_bounds__(256) void ln_to_bf16(const float* __restrict__ x,
                                                  const float* __restrict__ w,
                                                  const float* __restrict__ b,
                                                  unsigned short* __restrict__ out) {
    __shared__ float red0[8], red1[8];
    const int row = blockIdx.x;
    const float* xr = x + (size_t)row * D_MODEL;
    float s = 0.f, s2 = 0.f;
    for (int i = threadIdx.x; i < D_MODEL; i += 256) {
        float v = xr[i];
        s += v; s2 += v * v;
    }
#pragma unroll
    for (int off = 16; off > 0; off >>= 1) {
        s  += __shfl_down(s,  off, 32);
        s2 += __shfl_down(s2, off, 32);
    }
    int wave = threadIdx.x >> 5, lane = threadIdx.x & 31;
    if (lane == 0) { red0[wave] = s; red1[wave] = s2; }
    __syncthreads();
    if (threadIdx.x == 0) {
        float a = 0.f, c = 0.f;
        for (int i = 0; i < 8; ++i) { a += red0[i]; c += red1[i]; }
        red0[0] = a; red1[0] = c;
    }
    __syncthreads();
    float mean = red0[0] * (1.f / D_MODEL);
    float var  = red1[0] * (1.f / D_MODEL) - mean * mean;
    float inv  = rsqrtf(var + 1e-5f);
    for (int i = threadIdx.x; i < D_MODEL; i += 256) {
        float v = (xr[i] - mean) * inv * w[i] + b[i];
        out[(size_t)row * D_MODEL + i] = f2b(v);
    }
}

// ---------------------------------------------------------------------------
// WMMA GEMM: D[M,N] = A[M,K](bf16) @ B[K,N](bf16). Block tile 64x64, 4 waves.
// A tile staged by the async global->LDS engine (double buffered, ASYNCcnt)
// when available, else register-pipelined. B staged transposed [n][k] with
// packed b32 stores, register-pipelined across the barriers.
// MODE 0: f32 (+bias)  1: bf16 (+bias)  2: bf16 gelu(+bias)  3: f32 resid+bias
// ---------------------------------------------------------------------------
template <int MODE, int N>
__global__ __launch_bounds__(128) void gemm_bf16(const unsigned short* __restrict__ A,
                                                 const unsigned short* __restrict__ B,
                                                 void* __restrict__ D,
                                                 const float* __restrict__ bias,
                                                 const float* __restrict__ resid,
                                                 int M, int K) {
    const int LDA = 40, LDB = 40;
    __shared__ __align__(16) unsigned short As[2][64 * LDA];
    __shared__ __align__(16) unsigned short BsT[64 * LDB];

    const int m0 = blockIdx.y * 64, n0 = blockIdx.x * 64;
    const int tid = threadIdx.x, wave = tid >> 5, lane = tid & 31;

    const int ar = tid >> 1, ac = (tid & 1) * 16;        // A: 64 rows x 32 cols
    const int brp = (tid >> 3) * 2, bc = (tid & 7) * 8;  // B: 16 row-pairs x 8 col-grps

    v8f acc[4];
#pragma unroll
    for (int t = 0; t < 4; ++t)
#pragma unroll
        for (int e = 0; e < 8; ++e) acc[t][e] = 0.f;

    const unsigned short* gA = A + (size_t)(m0 + ar) * K + ac;
    const unsigned short* gB = B + (size_t)brp * N + n0 + bc;

    // pipeline preload of first K-tile
#if HAVE_ASYNC_LDS
    async_cp16(&As[0][ar * LDA + ac],     gA);
    async_cp16(&As[0][ar * LDA + ac + 8], gA + 8);
    us8 a0{}, a1{};  // unused in async mode
#else
    us8 a0 = *(const us8*)gA;
    us8 a1 = *(const us8*)(gA + 8);
#endif
    us8 r0 = *(const us8*)gB;
    us8 r1 = *(const us8*)(gB + N);

    for (int kk = 0; kk < K; kk += 32) {
        const int cur = (kk >> 5) & 1;
#if !HAVE_ASYNC_LDS
        *(us8*)&As[0][ar * LDA + ac]     = a0;
        *(us8*)&As[0][ar * LDA + ac + 8] = a1;
#endif
#pragma unroll
        for (int i = 0; i < 8; ++i) {
            unsigned int pk = (unsigned int)r0[i] | ((unsigned int)r1[i] << 16);
            *(unsigned int*)&BsT[(bc + i) * LDB + brp] = pk;
        }
#if HAVE_ASYNC_LDS
        __builtin_amdgcn_s_wait_asynccnt(0);  // this tile's A landed in LDS
#endif
        __syncthreads();

        // issue next tile's loads; results consumed after this wmma group
        int kn = (kk + 32 < K) ? kk + 32 : kk;
        const unsigned short* gA2 = A + (size_t)(m0 + ar) * K + kn + ac;
        const unsigned short* gB2 = B + (size_t)(kn + brp) * N + n0 + bc;
#if HAVE_ASYNC_LDS
        async_cp16(&As[1 - cur][ar * LDA + ac],     gA2);
        async_cp16(&As[1 - cur][ar * LDA + ac + 8], gA2 + 8);
#else
        us8 na0 = *(const us8*)gA2;
        us8 na1 = *(const us8*)(gA2 + 8);
#endif
        us8 nr0 = *(const us8*)gB2;
        us8 nr1 = *(const us8*)(gB2 + N);
        if (kk + 64 < K) {
            __builtin_prefetch(A + (size_t)(m0 + ar) * K + kk + 64 + ac, 0, 1);
            __builtin_prefetch(B + (size_t)(kk + 64 + brp) * N + n0 + bc, 0, 1);
        }

        const unsigned short* Asel = As[HAVE_ASYNC_LDS ? cur : 0];
        v16bf af = load_a_vec(&Asel[(wave * 16) * LDA], LDA, lane);
#pragma unroll
        for (int nt = 0; nt < 4; ++nt) {
            v16bf bf_ = load_b_vecT(&BsT[(nt * 16) * LDB], LDB, lane);
            acc[nt] = wmma_bf16(af, bf_, acc[nt]);
        }
        __syncthreads();
#if !HAVE_ASYNC_LDS
        a0 = na0; a1 = na1;
#endif
        r0 = nr0; r1 = nr1;
    }

    const int n_l = lane & 15;
    const int mhi = (lane & 16) ? 8 : 0;
    const int gmBase = m0 + wave * 16 + mhi;
#pragma unroll
    for (int nt = 0; nt < 4; ++nt) {
        int gn = n0 + nt * 16 + n_l;
        float bv = bias ? bias[gn] : 0.f;
        size_t base = (size_t)gmBase * N + gn;
        if (MODE == 0) {
            float* p = (float*)D + base;
#pragma unroll
            for (int r = 0; r < 8; ++r) p[r * N] = acc[nt][r] + bv;
        } else if (MODE == 3) {
            float* p = (float*)D + base;
            const float* rp = resid + base;
#pragma unroll
            for (int r = 0; r < 8; ++r) p[r * N] = acc[nt][r] + bv + rp[r * N];
        } else {
            unsigned short* p = (unsigned short*)D + base;
#pragma unroll
            for (int r = 0; r < 8; ++r) {
                float v = acc[nt][r] + bv;
                if (MODE == 2) v = 0.5f * v * (1.f + erff(v * 0.70710678118654752f));
                p[r * N] = f2b(v);
            }
        }
    }
}

// ---------------------------------------------------------------------------
// l2-normalize per (token, head) group of 64 fp32 -> bf16
// ---------------------------------------------------------------------------
__global__ __launch_bounds__(64) void l2norm64(const float* __restrict__ in,
                                               unsigned short* __restrict__ out) {
    __shared__ float sm[2];
    size_t base = (size_t)blockIdx.x * 64;
    float v = in[base + threadIdx.x];
    float sq = v * v;
#pragma unroll
    for (int off = 16; off > 0; off >>= 1) sq += __shfl_xor(sq, off, 32);
    int wave = threadIdx.x >> 5;
    if ((threadIdx.x & 31) == 0) sm[wave] = sq;
    __syncthreads();
    float nrm = fmaxf(sqrtf(sm[0] + sm[1]), 1e-12f);
    out[base + threadIdx.x] = f2b(v / nrm);
}

__global__ void sigmoid_ip(float* __restrict__ t, size_t n) {
    for (size_t i = blockIdx.x * (size_t)blockDim.x + threadIdx.x; i < n;
         i += (size_t)gridDim.x * blockDim.x)
        t[i] = 1.f / (1.f + __expf(-t[i]));
}

// ---------------------------------------------------------------------------
// Chunked GLA scan. One block per (batch, head), 8 waves, 64 chunks serial.
// Dynamic LDS (72192 B):
//   qS [t][d] us72 | kS [t][d] us72 | vT [e][t] us72 | SbT [e][f] us72 (bf16 S)
//   laS f68 (la_cum; aliased later by attn us72 + kdT us64) | SstT [e][f] f68
//   lam[64] | cdv[64]
// Next chunk's global data is preloaded into registers during compute.
// ---------------------------------------------------------------------------
#define LDQ 72
#define LAF 68
#define GLA_SMEM (4 * 64 * LDQ * 2 + 2 * 64 * LAF * 4 + 2 * 64 * 4)

__global__ __launch_bounds__(256) void gla_kernel(const unsigned short* __restrict__ qb,
                                                  const unsigned short* __restrict__ kb,
                                                  const unsigned short* __restrict__ vb,
                                                  const float* __restrict__ alpha,
                                                  unsigned short* __restrict__ ob) {
    extern __shared__ __align__(16) unsigned char smem[];
    unsigned short* qS  = (unsigned short*)smem;              // 64*72
    unsigned short* kS  = qS + 64 * LDQ;                      // 64*72
    unsigned short* vT  = kS + 64 * LDQ;                      // 64*72
    unsigned short* SbT = vT + 64 * LDQ;                      // 64*72 bf16 state
    float* laS  = (float*)(SbT + 64 * LDQ);                   // 64*68 f32
    float* SstT = laS + 64 * LAF;                             // 64*68 f32 state
    float* lam  = SstT + 64 * LAF;                            // 64
    float* cdv  = lam + 64;                                   // 64
    unsigned short* atS = (unsigned short*)laS;               // alias: attn, ld 72
    unsigned short* kdT = (unsigned short*)laS + 64 * LDQ;    // alias: kd^T, ld 64

    const int bh = blockIdx.x;
    const int b = bh >> 4, h = bh & 15;
    const int tid = threadIdx.x, wave = tid >> 5, lane = tid & 31;
    const int n_l = lane & 15, mhi = (lane & 16) ? 8 : 0;
    const size_t headcol = (size_t)h * DHEAD;

    // staging coords: thread owns 2 consecutive rows x 8 cols
    const int tp = (tid >> 3) * 2, lc = (tid & 7) * 8;

    for (int e = tid; e < 64 * 64; e += 256) {
        SstT[(e >> 6) * LAF + (e & 63)] = 0.f;
        SbT[(e >> 6) * LDQ + (e & 63)]  = 0;
    }
    __syncthreads();

    // chunk-data pipeline registers
    us8 rq0, rq1, rk0, rk1, rv0, rv1;
    f4  ra0, ra1, ra2, ra3;
    auto load_regs = [&](int t0x) {
        size_t g0 = ((size_t)(b * SEQ_N + t0x + tp)) * D_MODEL + headcol + lc;
        size_t g1 = g0 + D_MODEL;
        rq0 = *(const us8*)(qb + g0); rq1 = *(const us8*)(qb + g1);
        rk0 = *(const us8*)(kb + g0); rk1 = *(const us8*)(kb + g1);
        rv0 = *(const us8*)(vb + g0); rv1 = *(const us8*)(vb + g1);
        ra0 = *(const f4*)(alpha + g0); ra1 = *(const f4*)(alpha + g0 + 4);
        ra2 = *(const f4*)(alpha + g1); ra3 = *(const f4*)(alpha + g1 + 4);
    };
    load_regs(0);

    for (int t0 = 0; t0 < SEQ_N; t0 += CHUNK) {
        // ---- phase 1: stage chunk from regs ----
        *(us8*)&qS[tp * LDQ + lc]       = rq0;
        *(us8*)&qS[(tp + 1) * LDQ + lc] = rq1;
        *(us8*)&kS[tp * LDQ + lc]       = rk0;
        *(us8*)&kS[(tp + 1) * LDQ + lc] = rk1;
#pragma unroll
        for (int i = 0; i < 8; ++i) {
            unsigned int pk = (unsigned int)rv0[i] | ((unsigned int)rv1[i] << 16);
            *(unsigned int*)&vT[(lc + i) * LDQ + tp] = pk;
        }
        *(f4*)&laS[tp * LAF + lc]           = ra0;
        *(f4*)&laS[tp * LAF + lc + 4]       = ra1;
        *(f4*)&laS[(tp + 1) * LAF + lc]     = ra2;
        *(f4*)&laS[(tp + 1) * LAF + lc + 4] = ra3;
        __syncthreads();

        // issue next chunk's global loads now; consumed next iteration
        load_regs((t0 + CHUNK < SEQ_N) ? t0 + CHUNK : t0);

        // ---- phase 2: log + cumsum along time (thread d owns a column) ----
        if (tid < 64) {
            float c = 0.f;
            for (int t = 0; t < 64; ++t) {
                c += logf(fmaxf(laS[t * LAF + tid], 1e-6f));
                laS[t * LAF + tid] = c;
            }
        }
        __syncthreads();

        // ---- phase 3: row means + chunk decay ----
        if (tid < 64) {
            float s = 0.f;
            for (int d = 0; d < 64; ++d) s += laS[tid * LAF + d];
            lam[tid] = s * (1.f / 64.f);
            cdv[tid] = __expf(laS[63 * LAF + tid]);
        }
        __syncthreads();

        // ---- phase 4a: attn = q @ k^T (accumulators in regs) ----
        v8f at[2];
        const int ti0 = wave * 2;
#pragma unroll
        for (int i = 0; i < 2; ++i) {
            int mi = (ti0 + i) >> 2, ni = (ti0 + i) & 3;
#pragma unroll
            for (int e = 0; e < 8; ++e) at[i][e] = 0.f;
#pragma unroll
            for (int ks = 0; ks < 64; ks += 32) {
                v16bf a  = load_a_vec(&qS[(mi * 16) * LDQ + ks], LDQ, lane);
                v16bf bb = load_b_vecT(&kS[(ni * 16) * LDQ + ks], LDQ, lane);
                at[i] = wmma_bf16(a, bb, at[i]);
            }
        }
        __syncthreads();

        // ---- phase 4b: q_dec in place (vectorized); k_dec staged packed in regs ----
        unsigned int kdpk[8];
        {
            f4 l63a = *(const f4*)&laS[63 * LAF + lc];
            f4 l63b = *(const f4*)&laS[63 * LAF + lc + 4];
            float l63[8] = {l63a[0], l63a[1], l63a[2], l63a[3],
                            l63b[0], l63b[1], l63b[2], l63b[3]};
            unsigned short kd0[8], kd1[8];
#pragma unroll
            for (int rr = 0; rr < 2; ++rr) {
                int t = tp + rr;
                f4 laA = *(const f4*)&laS[t * LAF + lc];
                f4 laB = *(const f4*)&laS[t * LAF + lc + 4];
                float la[8] = {laA[0], laA[1], laA[2], laA[3],
                               laB[0], laB[1], laB[2], laB[3]};
                us8 qv = *(const us8*)&qS[t * LDQ + lc];
                us8 kv = *(const us8*)&kS[t * LDQ + lc];
                us8 qo;
#pragma unroll
                for (int i = 0; i < 8; ++i) {
                    qo[i] = f2b(b2f(qv[i]) * __expf(la[i]));
                    unsigned short kd = f2b(b2f(kv[i]) * __expf(l63[i] - la[i]));
                    if (rr == 0) kd0[i] = kd; else kd1[i] = kd;
                }
                *(us8*)&qS[t * LDQ + lc] = qo;
            }
#pragma unroll
            for (int i = 0; i < 8; ++i)
                kdpk[i] = (unsigned int)kd0[i] | ((unsigned int)kd1[i] << 16);
        }
        __syncthreads();

        // ---- phase 4c: la_cum dead; write attn(bf16) and kd^T into its space ----
#pragma unroll
        for (int i = 0; i < 8; ++i)
            *(unsigned int*)&kdT[(lc + i) * 64 + tp] = kdpk[i];
#pragma unroll
        for (int i = 0; i < 2; ++i) {
            int mi = (ti0 + i) >> 2, ni = (ti0 + i) & 3;
#pragma unroll
            for (int r = 0; r < 8; ++r) {
                int m = mi * 16 + r + mhi;
                int n = ni * 16 + n_l;
                float dcy = (n <= m) ? __expf(lam[m] - lam[n]) : 0.f;
                atS[m * LDQ + n] = f2b(at[i][r] * dcy);
            }
        }
        __syncthreads();

        // ---- phase 5: o = attn @ v + q_dec @ S (bf16 shadow) ----
#pragma unroll
        for (int i = 0; i < 2; ++i) {
            int mi = (ti0 + i) >> 2, ni = (ti0 + i) & 3;
            v8f oc;
#pragma unroll
            for (int e = 0; e < 8; ++e) oc[e] = 0.f;
#pragma unroll
            for (int ks = 0; ks < 64; ks += 32) {
                v16bf a  = load_a_vec(&atS[(mi * 16) * LDQ + ks], LDQ, lane);
                v16bf bb = load_b_vecT(&vT[(ni * 16) * LDQ + ks], LDQ, lane);
                oc = wmma_bf16(a, bb, oc);
            }
#pragma unroll
            for (int ks = 0; ks < 64; ks += 32) {
                v16bf a  = load_a_vec(&qS[(mi * 16) * LDQ + ks], LDQ, lane);
                v16bf bb = load_b_vecT(&SbT[(ni * 16) * LDQ + ks], LDQ, lane);
                oc = wmma_bf16(a, bb, oc);
            }
            unsigned short* po = ob +
                ((size_t)(b * SEQ_N + t0 + mi * 16 + mhi)) * D_MODEL + headcol +
                ni * 16 + n_l;
#pragma unroll
            for (int r = 0; r < 8; ++r) po[r * D_MODEL] = f2b(oc[r]);
        }
        __syncthreads();

        // ---- phase 6: kv = kd^T @ v ; S = S*cd + kv (f32 + bf16 shadow) ----
#pragma unroll
        for (int i = 0; i < 2; ++i) {
            int mi = (ti0 + i) >> 2, ni = (ti0 + i) & 3;
            v8f kv;
#pragma unroll
            for (int e = 0; e < 8; ++e) kv[e] = 0.f;
#pragma unroll
            for (int ks = 0; ks < 64; ks += 32) {
                v16bf a  = load_a_vec(&kdT[(mi * 16) * 64 + ks], 64, lane);
                v16bf bb = load_b_vecT(&vT[(ni * 16) * LDQ + ks], LDQ, lane);
                kv = wmma_bf16(a, bb, kv);
            }
            int e  = ni * 16 + n_l;
            int f0 = mi * 16 + mhi;
            f4 s0 = *(const f4*)&SstT[e * LAF + f0];
            f4 s1 = *(const f4*)&SstT[e * LAF + f0 + 4];
            f4 c0 = *(const f4*)&cdv[f0];
            f4 c1 = *(const f4*)&cdv[f0 + 4];
            f4 n0v, n1v;
            us8 sb;
#pragma unroll
            for (int r = 0; r < 4; ++r) {
                n0v[r] = s0[r] * c0[r] + kv[r];
                n1v[r] = s1[r] * c1[r] + kv[r + 4];
                sb[r]     = f2b(n0v[r]);
                sb[r + 4] = f2b(n1v[r]);
            }
            *(f4*)&SstT[e * LAF + f0]     = n0v;
            *(f4*)&SstT[e * LAF + f0 + 4] = n1v;
            *(us8*)&SbT[e * LDQ + f0]     = sb;
        }
        __syncthreads();
    }
}

// ---------------------------------------------------------------------------
// Host-side launch
// ---------------------------------------------------------------------------
extern "C" void kernel_launch(void* const* d_in, const int* in_sizes, int n_in,
                              void* d_out, int out_size, void* d_ws, size_t ws_size,
                              hipStream_t stream) {
    const float* x     = (const float*)d_in[0];
    const float* Wq    = (const float*)d_in[1];
    const float* Wk    = (const float*)d_in[2];
    const float* Wv    = (const float*)d_in[3];
    const float* Wg    = (const float*)d_in[4];
    const float* bg    = (const float*)d_in[5];
    const float* Wo    = (const float*)d_in[6];
    const float* ln1_w = (const float*)d_in[7];
    const float* ln1_b = (const float*)d_in[8];
    const float* ln2_w = (const float*)d_in[9];
    const float* ln2_b = (const float*)d_in[10];
    const float* W1    = (const float*)d_in[11];
    const float* b1    = (const float*)d_in[12];
    const float* W2    = (const float*)d_in[13];
    const float* b2    = (const float*)d_in[14];
    float* out = (float*)d_out;

    char* ws = (char*)d_ws;
    size_t off = 0;
    auto alloc = [&](size_t bytes) -> void* {
        off = (off + 255) & ~(size_t)255;
        void* p = ws + off;
        off += bytes;
        return p;
    };
    const size_t SQ  = (size_t)D_MODEL * D_MODEL;
    const size_t ACT = (size_t)TOKENS * D_MODEL;
    unsigned short* wq_b = (unsigned short*)alloc(SQ * 2);
    unsigned short* wk_b = (unsigned short*)alloc(SQ * 2);
    unsigned short* wv_b = (unsigned short*)alloc(SQ * 2);
    unsigned short* wg_b = (unsigned short*)alloc(SQ * 2);
    unsigned short* wo_b = (unsigned short*)alloc(SQ * 2);
    unsigned short* w1_b = (unsigned short*)alloc((size_t)D_MODEL * FFN_DIM * 2);
    unsigned short* w2_b = (unsigned short*)alloc((size_t)FFN_DIM * D_MODEL * 2);
    unsigned short* xn_b = (unsigned short*)alloc(ACT * 2);
    float*          tmpF = (float*)alloc(ACT * 4);
    unsigned short* q_b  = (unsigned short*)alloc(ACT * 2);
    unsigned short* k_b  = (unsigned short*)alloc(ACT * 2);
    unsigned short* v_b  = (unsigned short*)alloc(ACT * 2);
    unsigned short* o_b  = (unsigned short*)alloc(ACT * 2);
    float*          x1   = (float*)alloc(ACT * 4);
    unsigned short* g1_b = (unsigned short*)alloc((size_t)TOKENS * FFN_DIM * 2);

    cvt_f32_bf16<<<2048, 256, 0, stream>>>(Wq, wq_b, SQ);
    cvt_f32_bf16<<<2048, 256, 0, stream>>>(Wk, wk_b, SQ);
    cvt_f32_bf16<<<2048, 256, 0, stream>>>(Wv, wv_b, SQ);
    cvt_f32_bf16<<<2048, 256, 0, stream>>>(Wg, wg_b, SQ);
    cvt_f32_bf16<<<2048, 256, 0, stream>>>(Wo, wo_b, SQ);
    cvt_f32_bf16<<<4096, 256, 0, stream>>>(W1, w1_b, (size_t)D_MODEL * FFN_DIM);
    cvt_f32_bf16<<<4096, 256, 0, stream>>>(W2, w2_b, (size_t)FFN_DIM * D_MODEL);

    ln_to_bf16<<<TOKENS, 256, 0, stream>>>(x, ln1_w, ln1_b, xn_b);

    dim3 gProj(D_MODEL / 64, TOKENS / 64);
    dim3 gFfn1(FFN_DIM / 64, TOKENS / 64);

    gemm_bf16<0, D_MODEL><<<gProj, 128, 0, stream>>>(xn_b, wq_b, tmpF, nullptr, nullptr,
                                                     TOKENS, D_MODEL);
    l2norm64<<<TOKENS * NHEAD, 64, 0, stream>>>(tmpF, q_b);

    gemm_bf16<0, D_MODEL><<<gProj, 128, 0, stream>>>(xn_b, wk_b, tmpF, nullptr, nullptr,
                                                     TOKENS, D_MODEL);
    l2norm64<<<TOKENS * NHEAD, 64, 0, stream>>>(tmpF, k_b);

    gemm_bf16<1, D_MODEL><<<gProj, 128, 0, stream>>>(xn_b, wv_b, v_b, nullptr, nullptr,
                                                     TOKENS, D_MODEL);

    gemm_bf16<0, D_MODEL><<<gProj, 128, 0, stream>>>(xn_b, wg_b, tmpF, bg, nullptr,
                                                     TOKENS, D_MODEL);
    sigmoid_ip<<<2048, 256, 0, stream>>>(tmpF, ACT);

    gla_kernel<<<BATCH * NHEAD, 256, GLA_SMEM, stream>>>(q_b, k_b, v_b, tmpF, o_b);

    gemm_bf16<3, D_MODEL><<<gProj, 128, 0, stream>>>(o_b, wo_b, x1, nullptr, x,
                                                     TOKENS, D_MODEL);

    ln_to_bf16<<<TOKENS, 256, 0, stream>>>(x1, ln2_w, ln2_b, xn_b);

    gemm_bf16<2, FFN_DIM><<<gFfn1, 128, 0, stream>>>(xn_b, w1_b, g1_b, b1, nullptr,
                                                     TOKENS, D_MODEL);

    gemm_bf16<3, D_MODEL><<<gProj, 128, 0, stream>>>(g1_b, w2_b, out, b2, x1,
                                                     TOKENS, FFN_DIM);
}
